// VisionTransformer_58531814310188
// MI455X (gfx1250) — matrix operand order repaired and checked by reading, not compiled, not run on hardware
//
#include <hip/hip_runtime.h>

// ---------------------------------------------------------------------------
// CDNA5 (gfx1250) Vision Transformer forward, v3.
// - bf16 operands pre-swizzled into WMMA fragment layout (convert once).
// - GEMM: 128x128 tile, double-buffered LDS, async global->LDS DMA
//   (global_load_async_to_lds_b128 / ASYNCcnt) overlapped with 8 WMMAs/wave.
// - Attention: one block per (b,h); K/V head-slice staged in 98.5KB dynamic
//   LDS, K/Q L2-normalized in-kernel (rownorm pass eliminated).
// ---------------------------------------------------------------------------

#ifndef USE_ASYNC
#define USE_ASYNC 1
#endif

typedef __attribute__((ext_vector_type(16))) __bf16 v16bf;
typedef __attribute__((ext_vector_type(8)))  float  v8f;

union Frag {
    v16bf v;
    uint4 q[2];
};

// Model dims
#define BSZ   32
#define TT    197
#define NPAT  196
#define DD    768
#define HH    12
#define DK    64
#define FFD   3072
#define NCLS  1000
#define MROWS (BSZ * TT)   // 6304 (= 394*16)
#define MPAT  (BSZ * NPAT) // 6272 (= 392*16)

__device__ __forceinline__ unsigned short to_bf16(float f) {
    __bf16 h = (__bf16)f;
    return __builtin_bit_cast(unsigned short, h);
}

__device__ __forceinline__ float gelu_exact(float x) {
    return 0.5f * x * (1.0f + erff(x * 0.70710678118654752440f));
}

// A-matrix (16x32 bf16, ISA 7.12.2) swizzled storage. Element (m,k), row len K.
__device__ __forceinline__ size_t a_sw_u16(int m, int k, int K) {
    const int mb   = m >> 4, kb = k >> 5;
    const int lane = (m & 15) + ((k & 8) ? 16 : 0);
    const int dw   = ((k & 16) ? 4 : 0) + ((k & 7) >> 1);
    return ((((size_t)mb * (K >> 5) + kb) * 32 + lane) * 8 + dw) * 2 + (k & 1);
}

// B-matrix (32x16 bf16) swizzled storage. Element (k,n).
__device__ __forceinline__ size_t b_sw_u16(int k, int n, int K) {
    const int nb   = n >> 4, kb = k >> 5;
    const int lane = (n & 15) + ((k & 8) ? 16 : 0);
    const int dw   = ((k & 16) ? 4 : 0) + ((k & 7) >> 1);
    return ((((size_t)nb * (K >> 5) + kb) * 32 + lane) * 8 + dw) * 2 + (k & 1);
}

// 32-byte per-lane global->LDS copy (one fragment slot).
__device__ __forceinline__ void copy32B(unsigned int* dst, const unsigned int* src) {
#if USE_ASYNC
    unsigned int       ldsoff = (unsigned int)(unsigned long long)(void*)dst;
    unsigned long long gaddr  = (unsigned long long)(const void*)src;
    asm volatile("global_load_async_to_lds_b128 %0, %1, off"
                 :: "v"(ldsoff), "v"(gaddr) : "memory");
    asm volatile("global_load_async_to_lds_b128 %0, %1, off offset:16"
                 :: "v"(ldsoff), "v"(gaddr) : "memory");
#else
    uint4 x0 = *(const uint4*)src;
    uint4 x1 = *((const uint4*)src + 1);
    *(uint4*)dst       = x0;
    *((uint4*)dst + 1) = x1;
#endif
}

__device__ __forceinline__ void loadFrag(Frag& f, const unsigned int* p) {
    f.q[0] = *(const uint4*)p;
    f.q[1] = *((const uint4*)p + 1);
}

// ---------------------------------------------------------------------------
// GEMM: C[M,N] = epi(Asw[M,K](bf16) @ Bsw[K,N](bf16))
// FUSE bit0: +bias; bit1: GELU; bit2: +res (f32, stride N)
// OUTSW 0: C = f32 [M,N]; 1: C = bf16 A-swizzled (row len N, N mult of 32)
// M multiple of 16; K multiple of 32. Double-buffered, async DMA pipeline.
// ---------------------------------------------------------------------------
template <int FUSE, int OUTSW>
__global__ __launch_bounds__(256)
void gemm_sw_wmma(const unsigned short* __restrict__ Aswz,
                  const unsigned short* __restrict__ Bswz,
                  const float* __restrict__ bias, const float* __restrict__ res,
                  void* __restrict__ Cv, int M, int N, int K)
{
    __shared__ __align__(16) unsigned int Af[2][8][32][8];
    __shared__ __align__(16) unsigned int Bf[2][8][32][8];

    const int tid  = threadIdx.x;
    const int wave = tid >> 5;
    const int lane = tid & 31;
    const int half = lane >> 4;
    const int l16  = lane & 15;

    const int KB   = K >> 5;
    const int MB16 = M >> 4;
    const int Np   = (N + 15) & ~15;
    const int NB16 = Np >> 4;

    const int mb0 = blockIdx.y * 8;
    const int nb0 = blockIdx.x * 8;
    const int m0  = mb0 << 4;
    const int n0  = nb0 << 4;

    const int wr = wave >> 1; // A subtiles 2*wr, 2*wr+1
    const int wc = wave & 1;  // B subtiles 4*wc .. 4*wc+3

    // Clamp staging indices: out-of-range blocks stage duplicate valid data;
    // results for those rows/cols are discarded by the epilogue guards.
    const int mbc = min(mb0 + wave, MB16 - 1);
    const int nbc = min(nb0 + wave, NB16 - 1);

    const unsigned int* srcA = (const unsigned int*)Aswz + ((size_t)mbc * KB) * 256 + lane * 8;
    const unsigned int* srcB = (const unsigned int*)Bswz + ((size_t)nbc * KB) * 256 + lane * 8;

    v8f acc[2][4] = {};

    // prologue: stage tile 0 into buffer 0
    copy32B(&Af[0][wave][lane][0], srcA);
    copy32B(&Bf[0][wave][lane][0], srcB);

    for (int kb = 0; kb < KB; ++kb) {
        const int cur = kb & 1;
        const int nxt = cur ^ 1;
#if USE_ASYNC
        asm volatile("s_wait_asynccnt 0x0" ::: "memory"); // own cur-tile DMAs done
#endif
        __syncthreads(); // cur visible everywhere; everyone done computing kb-1

        if (kb + 1 < KB) { // overlap next tile's DMA with this tile's WMMAs
            copy32B(&Af[nxt][wave][lane][0], srcA + (size_t)(kb + 1) * 256);
            copy32B(&Bf[nxt][wave][lane][0], srcB + (size_t)(kb + 1) * 256);
        }

        Frag a0, a1, b0, b1, b2, b3;
        loadFrag(a0, &Af[cur][2 * wr + 0][lane][0]);
        loadFrag(a1, &Af[cur][2 * wr + 1][lane][0]);
        loadFrag(b0, &Bf[cur][4 * wc + 0][lane][0]);
        loadFrag(b1, &Bf[cur][4 * wc + 1][lane][0]);
        loadFrag(b2, &Bf[cur][4 * wc + 2][lane][0]);
        loadFrag(b3, &Bf[cur][4 * wc + 3][lane][0]);

        acc[0][0] = __builtin_amdgcn_wmma_f32_16x16x32_bf16(false, a0.v, false, b0.v, (short)0, acc[0][0], false, false);
        acc[0][1] = __builtin_amdgcn_wmma_f32_16x16x32_bf16(false, a0.v, false, b1.v, (short)0, acc[0][1], false, false);
        acc[0][2] = __builtin_amdgcn_wmma_f32_16x16x32_bf16(false, a0.v, false, b2.v, (short)0, acc[0][2], false, false);
        acc[0][3] = __builtin_amdgcn_wmma_f32_16x16x32_bf16(false, a0.v, false, b3.v, (short)0, acc[0][3], false, false);
        acc[1][0] = __builtin_amdgcn_wmma_f32_16x16x32_bf16(false, a1.v, false, b0.v, (short)0, acc[1][0], false, false);
        acc[1][1] = __builtin_amdgcn_wmma_f32_16x16x32_bf16(false, a1.v, false, b1.v, (short)0, acc[1][1], false, false);
        acc[1][2] = __builtin_amdgcn_wmma_f32_16x16x32_bf16(false, a1.v, false, b2.v, (short)0, acc[1][2], false, false);
        acc[1][3] = __builtin_amdgcn_wmma_f32_16x16x32_bf16(false, a1.v, false, b3.v, (short)0, acc[1][3], false, false);
    }

    // ---- epilogue. D layout: VGPR r -> M = r + 8*half, N = l16 ----
#pragma unroll
    for (int i = 0; i < 2; ++i) {
#pragma unroll
        for (int j = 0; j < 4; ++j) {
#pragma unroll
            for (int r = 0; r < 8; ++r) {
                const int gm = m0 + (2 * wr + i) * 16 + r + 8 * half;
                const int gn = n0 + (4 * wc + j) * 16 + l16;
                if (gm < M && gn < N) {
                    float v = acc[i][j][r];
                    if (FUSE & 1) v += bias[gn];
                    if (FUSE & 2) v = gelu_exact(v);
                    if (FUSE & 4) v += res[(size_t)gm * N + gn];
                    if (OUTSW) {
                        ((unsigned short*)Cv)[a_sw_u16(gm, gn, N)] = to_bf16(v);
                    } else {
                        ((float*)Cv)[(size_t)gm * N + gn] = v;
                    }
                }
            }
        }
    }
}

// ---------------------------------------------------------------------------
// LayerNorm: one wave per 768-wide row; writes bf16 A-swizzled output.
// ---------------------------------------------------------------------------
__global__ __launch_bounds__(256)
void layernorm_sw_kernel(const float* __restrict__ X, const float* __restrict__ g,
                         const float* __restrict__ b, unsigned short* __restrict__ Y,
                         int rows)
{
    const int wave = threadIdx.x >> 5;
    const int lane = threadIdx.x & 31;
    const int row  = blockIdx.x * 8 + wave;
    if (row >= rows) return;

    const float* x = X + (size_t)row * DD;
    float v[24];
    float s = 0.0f;
#pragma unroll
    for (int j = 0; j < 24; ++j) { v[j] = x[lane + 32 * j]; s += v[j]; }
#pragma unroll
    for (int off = 16; off > 0; off >>= 1) s += __shfl_xor(s, off, 32);
    const float mean = s * (1.0f / (float)DD);

    float q = 0.0f;
#pragma unroll
    for (int j = 0; j < 24; ++j) { const float d = v[j] - mean; q += d * d; }
#pragma unroll
    for (int off = 16; off > 0; off >>= 1) q += __shfl_xor(q, off, 32);
    const float rinv = rsqrtf(q * (1.0f / (float)DD) + 1e-5f);

#pragma unroll
    for (int j = 0; j < 24; ++j) {
        const int c = lane + 32 * j;
        const float o = g[c] * (v[j] - mean) * rinv + b[c];
        Y[a_sw_u16(row, c, DD)] = to_bf16(o);
    }
}

// ---------------------------------------------------------------------------
// Fused angular attention: one block per (b,h). K/V head-slice staged in
// dynamic LDS (2 x 197 x 64 f32 = 98.5KB); K rows L2-normalized in LDS,
// Q normalized in registers. One wave per query row, 8 rows in flight.
// ---------------------------------------------------------------------------
__global__ __launch_bounds__(256)
void attention_fused_kernel(const float* __restrict__ Qf, const float* __restrict__ Kf,
                            const float* __restrict__ Vf, unsigned short* __restrict__ O)
{
    extern __shared__ float smem[];
    float* Ks = smem;            // [197][64]
    float* Vs = smem + TT * DK;  // [197][64]

    const int tid  = threadIdx.x;
    const int wave = tid >> 5;
    const int lane = tid & 31;
    const int h = blockIdx.x;
    const int b = blockIdx.y;

    // stage raw K/V head-slices (coalesced 64-float rows)
    const size_t base = ((size_t)b * TT) * DD + (size_t)h * DK;
    for (int i = tid; i < TT * DK; i += 256) {
        const int r = i >> 6;
        const int c = i & 63;
        Ks[i] = Kf[base + (size_t)r * DD + c];
        Vs[i] = Vf[base + (size_t)r * DD + c];
    }
    __syncthreads();

    // L2-normalize K rows in LDS (wave per row)
    for (int r = wave; r < TT; r += 8) {
        const float k0 = Ks[r * DK + lane];
        const float k1 = Ks[r * DK + 32 + lane];
        float ss = k0 * k0 + k1 * k1;
#pragma unroll
        for (int off = 16; off > 0; off >>= 1) ss += __shfl_xor(ss, off, 32);
        const float sc = 1.0f / fmaxf(sqrtf(ss), 1e-12f);
        Ks[r * DK + lane]      = k0 * sc;
        Ks[r * DK + 32 + lane] = k1 * sc;
    }
    __syncthreads();

    // one wave per query row
    for (int t = wave; t < TT; t += 8) {
        const int    mrow = b * TT + t;
        const size_t rowT = (size_t)mrow * DD + (size_t)h * DK;
        float q0 = Qf[rowT + lane];
        float q1 = Qf[rowT + 32 + lane];
        float qq = q0 * q0 + q1 * q1;
#pragma unroll
        for (int off = 16; off > 0; off >>= 1) qq += __shfl_xor(qq, off, 32);
        const float qs = 1.0f / fmaxf(sqrtf(qq), 1e-12f);
        q0 *= qs; q1 *= qs;

        float acc0 = 0.0f, acc1 = 0.0f, wsum = 0.0f;
        for (int s = 0; s < TT; ++s) {
            float p = q0 * Ks[s * DK + lane] + q1 * Ks[s * DK + 32 + lane];
#pragma unroll
            for (int off = 16; off > 0; off >>= 1) p += __shfl_xor(p, off, 32);
            const float sim = fminf(fmaxf(p, -0.999f), 0.999f);
            const float sc  = 1.0f - acosf(sim) * 0.31830988618379067f; // 1/pi
            const float m   = fmaxf(sc, 1e-6f);
            const float m2 = m * m, m4 = m2 * m2;
            const float w  = m4 * m4; // score^8
            wsum += w;
            acc0 += w * Vs[s * DK + lane];
            acc1 += w * Vs[s * DK + 32 + lane];
        }
        const float inv = 1.0f / (wsum + 1e-6f);
        O[a_sw_u16(mrow, h * DK + lane, DD)]      = to_bf16(acc0 * inv);
        O[a_sw_u16(mrow, h * DK + 32 + lane, DD)] = to_bf16(acc1 * inv);
    }
}

// ---------------------------------------------------------------------------
// Patchify: x[B,C,224,224] -> bf16 A-swizzled patches [6272, 768]
// ---------------------------------------------------------------------------
__global__ __launch_bounds__(256)
void patchify_sw_kernel(const float* __restrict__ x, unsigned short* __restrict__ P)
{
    const long long idx = (long long)blockIdx.x * 256 + threadIdx.x;
    if (idx >= (long long)MPAT * DD) return;
    const int d  = (int)(idx % DD);
    const int r  = (int)(idx / DD);
    const int b  = r / NPAT;
    const int np = r % NPAT;
    const int c  = d >> 8;
    const int rem = d & 255;
    const int py = rem >> 4;
    const int px = rem & 15;
    const int gy = np / 14;
    const int gx = np % 14;
    const float v = x[(((size_t)b * 3 + c) * 224 + gy * 16 + py) * 224 + gx * 16 + px];
    P[a_sw_u16(r, d, DD)] = to_bf16(v);
}

// tok[b,0] = cls + pos[0]; tok[b,1+p] = ptok[b,p] + pos[1+p]  (f32 residual base)
__global__ __launch_bounds__(256)
void assemble_kernel(const float* __restrict__ ptok, const float* __restrict__ cls,
                     const float* __restrict__ pos, float* __restrict__ X)
{
    const long long idx = (long long)blockIdx.x * 256 + threadIdx.x;
    if (idx >= (long long)MROWS * DD) return;
    const int d = (int)(idx % DD);
    const int r = (int)(idx / DD);
    const int b = r / TT;
    const int t = r % TT;
    const float v = (t == 0) ? cls[d] : ptok[((size_t)b * NPAT + (t - 1)) * DD + d];
    X[idx] = v + pos[(size_t)t * DD + d];
}

// gather cls rows (swizzled -> swizzled, 32x768)
__global__ __launch_bounds__(256)
void gather_cls_sw_kernel(const unsigned short* __restrict__ Xn,
                          unsigned short* __restrict__ Cb)
{
    const int idx = blockIdx.x * 256 + threadIdx.x;
    if (idx >= BSZ * DD) return;
    const int b = idx / DD;
    const int k = idx % DD;
    Cb[a_sw_u16(b, k, DD)] = Xn[a_sw_u16(b * TT, k, DD)];
}

// ---------------------------------------------------------------------------
// Weight conversion: f32 [layers,K,N] -> bf16 B-swizzled [layers,K,Np]
// ---------------------------------------------------------------------------
__global__ __launch_bounds__(256)
void weight_swz_kernel(const float* __restrict__ W, unsigned short* __restrict__ dst,
                       int K, int N, int Np, int layers)
{
    const long long idx = (long long)blockIdx.x * 256 + threadIdx.x;
    const long long per = (long long)K * Np;
    if (idx >= per * layers) return;
    const int l = (int)(idx / per);
    const long long rem = idx % per;
    const int k = (int)(rem / Np);
    const int n = (int)(rem % Np);
    const float v = (n < N) ? W[(size_t)l * K * N + (size_t)k * N + n] : 0.0f;
    dst[(size_t)l * K * Np + b_sw_u16(k, n, K)] = to_bf16(v);
}

// ---------------------------------------------------------------------------
// Host orchestration
// ---------------------------------------------------------------------------
static inline dim3 gemm_grid(int M, int N) {
    const int Np = (N + 15) & ~15;
    return dim3((unsigned)((Np + 127) / 128), (unsigned)((M + 127) / 128));
}

extern "C" void kernel_launch(void* const* d_in, const int* in_sizes, int n_in,
                              void* d_out, int out_size, void* d_ws, size_t ws_size,
                              hipStream_t stream)
{
    (void)in_sizes; (void)n_in; (void)out_size; (void)ws_size;

    const float* x     = (const float*)d_in[0];
    const float* Wp    = (const float*)d_in[1];
    const float* bp    = (const float*)d_in[2];
    const float* cls   = (const float*)d_in[3];
    const float* pos   = (const float*)d_in[4];
    const float* Wq    = (const float*)d_in[5];
    const float* bq    = (const float*)d_in[6];
    const float* Wk    = (const float*)d_in[7];
    const float* bk    = (const float*)d_in[8];
    const float* Wv    = (const float*)d_in[9];
    const float* bv    = (const float*)d_in[10];
    const float* Wo    = (const float*)d_in[11];
    const float* bo    = (const float*)d_in[12];
    const float* ln1_g = (const float*)d_in[13];
    const float* ln1_b = (const float*)d_in[14];
    const float* ln2_g = (const float*)d_in[15];
    const float* ln2_b = (const float*)d_in[16];
    const float* W1    = (const float*)d_in[17];
    const float* b1    = (const float*)d_in[18];
    const float* W2    = (const float*)d_in[19];
    const float* b2    = (const float*)d_in[20];
    const float* lnf_g = (const float*)d_in[21];
    const float* lnf_b = (const float*)d_in[22];
    const float* Wh    = (const float*)d_in[23];
    const float* bh    = (const float*)d_in[24];

    const size_t MD = (size_t)MROWS * DD;

    // f32 region
    float* X  = (float*)d_ws;     // residual base
    float* R  = X + MD;           // post-attention residual
    float* Qf = R + MD;           // Q / patch-embed out
    float* Kf = Qf + MD;
    float* Vf = Kf + MD;
    // bf16 (u16) region
    unsigned short* Abf = (unsigned short*)(Vf + MD);     // swizzled activations [6304,768]
    unsigned short* Hbf = Abf + MD;                       // swizzled hidden [6304,3072] / cls buf
    unsigned short* WpS = Hbf + (size_t)MROWS * FFD;
    unsigned short* WqS = WpS + (size_t)DD * DD;
    unsigned short* WkS = WqS + (size_t)4 * DD * DD;
    unsigned short* WvS = WkS + (size_t)4 * DD * DD;
    unsigned short* WoS = WvS + (size_t)4 * DD * DD;
    unsigned short* W1S = WoS + (size_t)4 * DD * DD;
    unsigned short* W2S = W1S + (size_t)4 * DD * FFD;
    unsigned short* WhS = W2S + (size_t)4 * FFD * DD;     // 768 x 1008 (padded)

    const dim3 blk(256);
    const int NpH = (NCLS + 15) & ~15; // 1008
    const unsigned attn_lds = (unsigned)(2 * TT * DK * sizeof(float)); // 100864 B

    // ---- weight pre-swizzle (bf16, B-fragment layout) ----
    {
        long long n;
        n = (long long)DD * DD;
        weight_swz_kernel<<<dim3((unsigned)((n + 255) / 256)), blk, 0, stream>>>(Wp, WpS, DD, DD, DD, 1);
        n = (long long)4 * DD * DD;
        weight_swz_kernel<<<dim3((unsigned)((n + 255) / 256)), blk, 0, stream>>>(Wq, WqS, DD, DD, DD, 4);
        weight_swz_kernel<<<dim3((unsigned)((n + 255) / 256)), blk, 0, stream>>>(Wk, WkS, DD, DD, DD, 4);
        weight_swz_kernel<<<dim3((unsigned)((n + 255) / 256)), blk, 0, stream>>>(Wv, WvS, DD, DD, DD, 4);
        weight_swz_kernel<<<dim3((unsigned)((n + 255) / 256)), blk, 0, stream>>>(Wo, WoS, DD, DD, DD, 4);
        n = (long long)4 * DD * FFD;
        weight_swz_kernel<<<dim3((unsigned)((n + 255) / 256)), blk, 0, stream>>>(W1, W1S, DD, FFD, FFD, 4);
        weight_swz_kernel<<<dim3((unsigned)((n + 255) / 256)), blk, 0, stream>>>(W2, W2S, FFD, DD, DD, 4);
        n = (long long)DD * NpH;
        weight_swz_kernel<<<dim3((unsigned)((n + 255) / 256)), blk, 0, stream>>>(Wh, WhS, DD, NCLS, NpH, 1);
    }

    // ---- patch embed ----
    {
        const long long n = (long long)MPAT * DD;
        patchify_sw_kernel<<<dim3((unsigned)((n + 255) / 256)), blk, 0, stream>>>(x, Abf);
        gemm_sw_wmma<1, 0><<<gemm_grid(MPAT, DD), blk, 0, stream>>>(Abf, WpS, bp, nullptr, Qf, MPAT, DD, DD);
        const long long n2 = (long long)MROWS * DD;
        assemble_kernel<<<dim3((unsigned)((n2 + 255) / 256)), blk, 0, stream>>>(Qf, cls, pos, X);
    }

    const dim3 gLN((MROWS + 7) / 8);
    const dim3 gAT(HH, BSZ);
    const dim3 gD  = gemm_grid(MROWS, DD);
    const dim3 gFF = gemm_grid(MROWS, FFD);

    for (int i = 0; i < 4; ++i) {
        const unsigned short* WqSi = WqS + (size_t)i * DD * DD;
        const unsigned short* WkSi = WkS + (size_t)i * DD * DD;
        const unsigned short* WvSi = WvS + (size_t)i * DD * DD;
        const unsigned short* WoSi = WoS + (size_t)i * DD * DD;
        const unsigned short* W1Si = W1S + (size_t)i * DD * FFD;
        const unsigned short* W2Si = W2S + (size_t)i * FFD * DD;

        layernorm_sw_kernel<<<gLN, blk, 0, stream>>>(X, ln1_g + (size_t)i * DD,
                                                     ln1_b + (size_t)i * DD, Abf, MROWS);
        gemm_sw_wmma<1, 0><<<gD, blk, 0, stream>>>(Abf, WqSi, bq + (size_t)i * DD, nullptr, Qf, MROWS, DD, DD);
        gemm_sw_wmma<1, 0><<<gD, blk, 0, stream>>>(Abf, WkSi, bk + (size_t)i * DD, nullptr, Kf, MROWS, DD, DD);
        gemm_sw_wmma<1, 0><<<gD, blk, 0, stream>>>(Abf, WvSi, bv + (size_t)i * DD, nullptr, Vf, MROWS, DD, DD);
        // fused: Q/K L2-normalize + angular attention (K/V staged in LDS)
        attention_fused_kernel<<<gAT, blk, attn_lds, stream>>>(Qf, Kf, Vf, Abf);
        // R = attn @ Wo + bo + X
        gemm_sw_wmma<5, 0><<<gD, blk, 0, stream>>>(Abf, WoSi, bo + (size_t)i * DD, X, R, MROWS, DD, DD);
        layernorm_sw_kernel<<<gLN, blk, 0, stream>>>(R, ln2_g + (size_t)i * DD,
                                                     ln2_b + (size_t)i * DD, Abf, MROWS);
        // hidden = gelu(ln2 @ W1 + b1), bf16 swizzled
        gemm_sw_wmma<3, 1><<<gFF, blk, 0, stream>>>(Abf, W1Si, b1 + (size_t)i * FFD, nullptr, Hbf, MROWS, FFD, DD);
        // X = hidden @ W2 + b2 + R
        gemm_sw_wmma<5, 0><<<gD, blk, 0, stream>>>(Hbf, W2Si, b2 + (size_t)i * DD, R, X, MROWS, DD, FFD);
    }

    // final LN -> Abf, gather cls -> Hbf (32x768 swizzled), head GEMM -> d_out
    layernorm_sw_kernel<<<gLN, blk, 0, stream>>>(X, lnf_g, lnf_b, Abf, MROWS);
    gather_cls_sw_kernel<<<dim3((BSZ * DD + 255) / 256), blk, 0, stream>>>(Abf, Hbf);
    gemm_sw_wmma<1, 0><<<gemm_grid(BSZ, NCLS), blk, 0, stream>>>(Hbf, WhS, bh, nullptr,
                                                                 (float*)d_out, BSZ, NCLS, DD);
}
